// MessagePassing_15212774163057
// MI455X (gfx1250) — matrix operand order repaired and checked by reading, not compiled, not run on hardware
//
#include <hip/hip_runtime.h>
#include <hip/hip_bf16.h>

typedef __attribute__((ext_vector_type(16))) _Float16 v16h;
typedef __attribute__((ext_vector_type(8)))  _Float16 v8h;
typedef __attribute__((ext_vector_type(8)))  float    v8f;
typedef __attribute__((ext_vector_type(4)))  unsigned int v4u;
typedef __attribute__((ext_vector_type(8)))  int      v8i;
typedef __attribute__((ext_vector_type(4)))  int      v4i;

#define B_  4
#define N_  512
#define FN_ 64
#define FE_ 32
#define H_  128

// ---------------------------------------------------------------------------
// Tensor Data Mover: build a D# (ISA cdna5 §8) and issue tensor_load_to_lds.
// 2-D tile: tile_d0 contiguous elements (4B), tile_d1 rows, row stride
// stride0 elements. tensor_dims set huge so no OOB clipping occurs.
// ---------------------------------------------------------------------------
__device__ static inline void tdm_load_2d(unsigned lds_off, const void* gptr,
                                          unsigned tile_d0, unsigned tile_d1,
                                          unsigned stride0) {
  unsigned long long ga = (unsigned long long)(uintptr_t)gptr;
  v4u g0 = { 1u,                                   // count=1 (valid), load, no gather
             lds_off,                              // lds_addr
             (unsigned)ga,                         // global_addr[31:0]
             (unsigned)((ga >> 32) & 0x01FFFFFFull) | (2u << 30) }; // addr[56:32] | type=2
  v8i g1 = { (int)(2u << 16),                      // data_size=4B, no multicast/pad/iterate
             0,                                    // abar=0, tensor_dim0[15:0]=0 (dim0=1<<20)
             0x10,                                 // tensor_dim0[31:16]=0x10, dim1[15:0]=0
             (int)((tile_d0 << 16) | 0x10u),       // tile_dim0 | tensor_dim1[31:16]=0x10
             (int)tile_d1,                         // tile_dim1 (tile_dim2=0)
             (int)stride0,                         // tensor_dim0_stride[31:0]
             0, 0 };
  v4i z4 = { 0, 0, 0, 0 };
#if __clang_major__ >= 23
  v8i z8 = { 0, 0, 0, 0, 0, 0, 0, 0 };
  __builtin_amdgcn_tensor_load_to_lds(g0, g1, z4, z4, z8, 0);
#else
  __builtin_amdgcn_tensor_load_to_lds(g0, g1, z4, z4, 0);
#endif
}

// ---------------------------------------------------------------------------
// prep: h = x @ W_node ; new_x initialized to h
// ---------------------------------------------------------------------------
__global__ void prep_h_kernel(const float* __restrict__ x,
                              const float* __restrict__ Wn,
                              float* __restrict__ h,
                              float* __restrict__ newx) {
  int id  = blockIdx.x * blockDim.x + threadIdx.x;   // 262144 threads
  int col = id & (H_ - 1);
  int bn  = id >> 7;
  const float* xr = x + (size_t)bn * FN_;
  float s = 0.f;
#pragma unroll 8
  for (int k = 0; k < FN_; ++k) s += xr[k] * Wn[k * H_ + col];
  h[id]    = s;
  newx[id] = s;
}

// hi = h @ W_xi + b_m1 ; hj = h @ W_xj
__global__ void prep_hij_kernel(const float* __restrict__ h,
                                const float* __restrict__ Wm1,
                                const float* __restrict__ bm1,
                                float* __restrict__ hi,
                                float* __restrict__ hj) {
  int id  = blockIdx.x * blockDim.x + threadIdx.x;
  int col = id & (H_ - 1);
  int bn  = id >> 7;
  const float* hr = h + (size_t)bn * H_;
  float si = bm1[col];
  float sj = 0.f;
#pragma unroll 8
  for (int k = 0; k < H_; ++k) {
    float hv = hr[k];
    si += hv * Wm1[k * H_ + col];
    sj += hv * Wm1[(H_ + k) * H_ + col];
  }
  hi[id] = si;
  hj[id] = sj;
}

// ---------------------------------------------------------------------------
// Pack f32 weights into f16 WMMA B-operand order (wave32):
//  operand(chunk c, coltile ct): lane l holds 16 halves, K = c*32 + (t/8)*16 + (l/16)*8 + t%8,
//  col = ct*16 + (l%16). Contiguous per lane -> one 32B LDS read later.
//  Layout: [We (1 chunk, 8 ct)] [We2 (4 chunks, 8 ct)] [Wm2 (4 chunks, 8 ct)]
// ---------------------------------------------------------------------------
__global__ void pack_weights_kernel(const float* __restrict__ We,
                                    const float* __restrict__ Wm1,
                                    const float* __restrict__ Wm2,
                                    _Float16* __restrict__ pW) {
  int id = blockIdx.x * blockDim.x + threadIdx.x;  // 36864 threads
  int local, krow_off = 0;
  const float* src;
  if (id < 4096)        { local = id;          src = We;  }
  else if (id < 20480)  { local = id - 4096;   src = Wm1; krow_off = 2 * H_; }  // W_e2 = W_m1[2H:3H]
  else                  { local = id - 20480;  src = Wm2; }
  int c    = local >> 12;
  int r    = local & 4095;
  int ct   = r >> 9;
  int r2   = r & 511;
  int lane = r2 >> 4;
  int t    = r2 & 15;
  int K    = c * 32 + ((t >> 3) << 4) + ((lane >> 4) << 3) + (t & 7);
  int col  = ct * 16 + (lane & 15);
  pW[id] = (_Float16)src[(size_t)(krow_off + K) * H_ + col];
}

// ---------------------------------------------------------------------------
// Fused main kernel
// grid: b(4) x i-tile(32) x j-split(8); block = 256 (8 waves)
// Per wave iteration: 16 pairs (i0..i0+15, j) = one WMMA M-tile, three WMMA
// stages (e, m, n); adj tiles arrive via TDM one iteration ahead.
// ---------------------------------------------------------------------------
#define LDS_WE    0            //  8 KB : W_edge packed
#define LDS_WE2   8192         // 32 KB : W_e2  packed
#define LDS_WM2   40960        // 32 KB : W_m2  packed
#define LDS_HIT   73728        //  8 KB : hi, transposed (col*16 + M)
#define LDS_HJ    81920        // 32 KB : hj rows for this j-range
#define LDS_STG   114688       // 32 KB : per-wave f16 staging (8 x 4KB)
#define LDS_ADJ   147456       // 16 KB : per-wave TDM adj tiles (8 x 2KB)
#define LDS_RED   81920        // 64 KB : reduction (overlays HJ+STG, post-loop)
#define LDS_TOTAL 163840       // = 320KB/2 -> two workgroups per WGP

__global__ __launch_bounds__(256) void mp_main_kernel(
    const float* __restrict__ adj,
    const float* __restrict__ adj_mask,
    const float* __restrict__ b_edge,
    const float* __restrict__ b_m2,
    const float* __restrict__ hi,
    const float* __restrict__ hj,
    const _Float16* __restrict__ pW,
    float* __restrict__ e_out,
    float* __restrict__ newx) {
  extern __shared__ char smem[];
  const int tid = threadIdx.x;
  const int wg  = blockIdx.x;
  const int b   = wg >> 8;
  const int it  = (wg >> 3) & 31;
  const int js  = wg & 7;
  const int i0  = it * 16;
  const int j0  = js * 64;

  const int w    = tid >> 5;
  const int lane = tid & 31;
  const int row  = lane & 15;
  const int grp  = lane >> 4;

  // ---- prefill: weights + hj via TDM (wave 0), hi transposed cooperatively
  if (w == 0) {
    tdm_load_2d(LDS_WE, pW, 18432, 1, 18432);                       // 72 KB weights
    tdm_load_2d(LDS_HJ, hj + ((size_t)b * N_ + j0) * H_, 8192, 1, 8192); // 32 KB hj
  }
  {
    float* dHIT = (float*)(smem + LDS_HIT);
    const float* sHI = hi + ((size_t)b * N_ + i0) * H_;
    for (int k = tid; k < 2048; k += 256) {
      int col = k >> 4, M = k & 15;
      dHIT[k] = sHI[M * H_ + col];
    }
  }
  if (w == 0) __builtin_amdgcn_s_wait_tensorcnt(0);
  __syncthreads();

  _Float16* stg = (_Float16*)(smem + LDS_STG) + w * (16 * H_);
  const float* dHIT = (const float*)(smem + LDS_HIT);
  const float* dHJ  = (const float*)(smem + LDS_HJ);
  const float* aL   = (const float*)(smem + LDS_ADJ + w * 2048);

  float eb[8], mb[8];
#pragma unroll
  for (int ct = 0; ct < 8; ++ct) {
    eb[ct] = b_edge[ct * 16 + row];
    mb[ct] = b_m2[ct * 16 + row];
  }

  v8f nb[8];
#pragma unroll
  for (int ct = 0; ct < 8; ++ct)
#pragma unroll
    for (int r = 0; r < 8; ++r) nb[ct][r] = 0.f;

  // issue TDM for this wave's first adj tile (16 rows x 32 floats, row stride N*FE)
  tdm_load_2d(LDS_ADJ + w * 2048,
              adj + (((size_t)b * N_ + i0) * N_ + (j0 + w)) * FE_,
              FE_, 16, N_ * FE_);

  for (int t = w; t < 64; t += 8) {
    const int j = j0 + t;

    // per-row masks (row M = grp*8 + r  ->  i = i0 + M)
    float mv[8];
    {
      const float* mrow = adj_mask + ((size_t)b * N_ + i0 + grp * 8) * N_ + j;
#pragma unroll
      for (int r = 0; r < 8; ++r) mv[r] = mrow[(size_t)r * N_];
    }

    // ---- A1: adj tile from LDS (TDM landed) -> f16 A-operand ----
    __builtin_amdgcn_s_wait_tensorcnt(0);
    v16h A1;
    {
      const float* abase = aL + row * FE_ + 8 * grp;
      v8f flo = *(const v8f*)(abase);
      v8f fhi = *(const v8f*)(abase + 16);
#pragma unroll
      for (int q = 0; q < 8; ++q) {
        A1[q]     = (_Float16)flo[q];
        A1[8 + q] = (_Float16)fhi[q];
      }
    }
    // all LDS reads of the tile are done -> pipeline next tile via TDM
    asm volatile("s_wait_dscnt 0x0" ::: "memory");
    if (t + 8 < 64) {
      tdm_load_2d(LDS_ADJ + w * 2048,
                  adj + (((size_t)b * N_ + i0) * N_ + (j + 8)) * FE_,
                  FE_, 16, N_ * FE_);
    }

    // ---- stage 1: e = adj @ W_edge + b_edge  (K=32 -> one WMMA per ct) ----
#pragma unroll
    for (int ct = 0; ct < 8; ++ct) {
      v16h Bw = *(const v16h*)(smem + LDS_WE + (size_t)(ct * 32 + lane) * 32);
      v8f acc = {};
      acc = __builtin_amdgcn_wmma_f32_16x16x32_f16(false, A1, false, Bw,
                                                   (short)0, acc, false, false);
      const int col = ct * 16 + row;
      float* erow = e_out + (((size_t)b * N_ + i0 + grp * 8) * N_ + j) * H_ + col;
#pragma unroll
      for (int r = 0; r < 8; ++r) {
        float ev = acc[r] + eb[ct];
        erow[(size_t)r * N_ * H_] = ev;                 // e output
        stg[(grp * 8 + r) * H_ + col] = (_Float16)ev;   // stage for next GEMM
      }
    }
    asm volatile("s_wait_dscnt 0x0" ::: "memory");

    // ---- A2: e tile in A layout: two contiguous 16B runs per chunk ----
    v16h A2[4];
#pragma unroll
    for (int c = 0; c < 4; ++c) {
      const _Float16* base = stg + row * H_ + c * 32 + 8 * grp;
      v8h lo = *(const v8h*)(base);
      v8h hh = *(const v8h*)(base + 16);
      A2[c] = __builtin_shufflevector(lo, hh, 0, 1, 2, 3, 4, 5, 6, 7,
                                      8, 9, 10, 11, 12, 13, 14, 15);
    }
    asm volatile("" ::: "memory");

    // ---- stage 2: m = relu(hi_i + hj_j + e @ W_e2 + b_m1) ----
#pragma unroll
    for (int ct = 0; ct < 8; ++ct) {
      const int col = ct * 16 + row;
      const float hjv = dHJ[t * H_ + col];
      v8f hv = *(const v8f*)(dHIT + col * 16 + grp * 8);
      v8f acc;
#pragma unroll
      for (int r = 0; r < 8; ++r) acc[r] = hv[r] + hjv;
#pragma unroll
      for (int c = 0; c < 4; ++c) {
        v16h Bw = *(const v16h*)(smem + LDS_WE2 +
                                 (size_t)((c * 8 + ct) * 32 + lane) * 32);
        acc = __builtin_amdgcn_wmma_f32_16x16x32_f16(false, A2[c], false, Bw,
                                                     (short)0, acc, false, false);
      }
#pragma unroll
      for (int r = 0; r < 8; ++r) {
        float m = acc[r];
        m = m > 0.f ? m : 0.f;
        stg[(grp * 8 + r) * H_ + col] = (_Float16)m;    // overwrite staging
      }
    }
    asm volatile("s_wait_dscnt 0x0" ::: "memory");

    // ---- A3: relu(m) in A layout ----
    v16h A3[4];
#pragma unroll
    for (int c = 0; c < 4; ++c) {
      const _Float16* base = stg + row * H_ + c * 32 + 8 * grp;
      v8h lo = *(const v8h*)(base);
      v8h hh = *(const v8h*)(base + 16);
      A3[c] = __builtin_shufflevector(lo, hh, 0, 1, 2, 3, 4, 5, 6, 7,
                                      8, 9, 10, 11, 12, 13, 14, 15);
    }
    asm volatile("" ::: "memory");

    // ---- stage 3: n = m @ W_m2 + b_m2, mask, accumulate over j ----
#pragma unroll
    for (int ct = 0; ct < 8; ++ct) {
      v8f acc = {};
#pragma unroll
      for (int c = 0; c < 4; ++c) {
        v16h Bw = *(const v16h*)(smem + LDS_WM2 +
                                 (size_t)((c * 8 + ct) * 32 + lane) * 32);
        acc = __builtin_amdgcn_wmma_f32_16x16x32_f16(false, A3[c], false, Bw,
                                                     (short)0, acc, false, false);
      }
#pragma unroll
      for (int r = 0; r < 8; ++r)
        nb[ct][r] += mv[r] * (acc[r] + mb[ct]);
    }
  }

  // ---- cross-wave reduction of neighbor sums, then atomicAdd into new_x ----
  __syncthreads();
  float* red = (float*)(smem + LDS_RED);
#pragma unroll
  for (int ct = 0; ct < 8; ++ct)
#pragma unroll
    for (int r = 0; r < 8; ++r)
      red[w * 2048 + (grp * 8 + r) * H_ + ct * 16 + row] = nb[ct][r];
  __syncthreads();
  for (int k = tid; k < 2048; k += 256) {
    float s = 0.f;
#pragma unroll
    for (int ww = 0; ww < 8; ++ww) s += red[ww * 2048 + k];
    int M = k >> 7, col = k & 127;
    atomicAdd(&newx[((size_t)b * N_ + i0 + M) * H_ + col], s);
  }
}

// ---------------------------------------------------------------------------
extern "C" void kernel_launch(void* const* d_in, const int* in_sizes, int n_in,
                              void* d_out, int out_size, void* d_ws, size_t ws_size,
                              hipStream_t stream) {
  const float* x        = (const float*)d_in[0];
  const float* adj      = (const float*)d_in[1];
  const float* adj_mask = (const float*)d_in[2];
  const float* W_node   = (const float*)d_in[3];
  const float* W_edge   = (const float*)d_in[4];
  const float* b_edge   = (const float*)d_in[5];
  const float* W_m1     = (const float*)d_in[6];
  const float* b_m1     = (const float*)d_in[7];
  const float* W_m2     = (const float*)d_in[8];
  const float* b_m2     = (const float*)d_in[9];

  float* out      = (float*)d_out;
  float* newx     = out;                                  // 4*512*128
  float* e_out    = out + (size_t)B_ * N_ * H_;           // 4*512*512*128
  float* mask_out = e_out + (size_t)B_ * N_ * N_ * H_;    // 4*512*512

  char*     ws = (char*)d_ws;
  float*    h  = (float*)ws;                    // 1 MB
  float*    hi = (float*)(ws + (1u << 20));     // 1 MB
  float*    hj = (float*)(ws + (2u << 20));     // 1 MB
  _Float16* pW = (_Float16*)(ws + (3u << 20));  // 72 KB packed weights

  hipMemcpyAsync(mask_out, adj_mask, (size_t)B_ * N_ * N_ * sizeof(float),
                 hipMemcpyDeviceToDevice, stream);

  prep_h_kernel<<<1024, 256, 0, stream>>>(x, W_node, h, newx);
  prep_hij_kernel<<<1024, 256, 0, stream>>>(h, W_m1, b_m1, hi, hj);
  pack_weights_kernel<<<144, 256, 0, stream>>>(W_edge, W_m1, W_m2, pW);
  mp_main_kernel<<<1024, 256, LDS_TOTAL, stream>>>(adj, adj_mask, b_edge, b_m2,
                                                   hi, hj, pW, e_out, newx);
}
